// SparseInstCriterion_71116068487983
// MI455X (gfx1250) — matrix operand order repaired and checked by reading, not compiled, not run on hardware
//
#include <hip/hip_runtime.h>
#include <hip/hip_bf16.h>
#include <math.h>

#define B_   4
#define N_   16
#define V_   32
#define HW_  25600
#define TOPK 7
#define EPS_DICE 1e-8f

#define TILES_PER_CHUNK 10      // 16 pixels per tile -> 160 px per chunk
#define CHUNKS          160     // 160 * 10 * 16 = 25600

typedef __attribute__((ext_vector_type(16))) _Float16 v16h;
typedef __attribute__((ext_vector_type(8)))  _Float16 v8h;
typedef __attribute__((ext_vector_type(8)))  float    v8f;

// ---------------------------------------------------------------- zero
__global__ void zero_kernel(float* out, float* acc, int accN) {
    if (blockIdx.x == 0 && threadIdx.x < 3) out[threadIdx.x] = 0.f;
    for (int i = blockIdx.x * blockDim.x + threadIdx.x; i < accN;
         i += gridDim.x * blockDim.x)
        acc[i] = 0.f;
}

// ------------------------------------------- feat(f32, b,v,p) -> featT(f16, b,p,v)
__global__ __launch_bounds__(256) void convert_kernel(const float* __restrict__ feat,
                                                      _Float16* __restrict__ featT) {
    int idx = blockIdx.x * blockDim.x + threadIdx.x;
    const int total = B_ * V_ * HW_;
    if (idx >= total) return;
    int b   = idx / (V_ * HW_);
    int rem = idx - b * (V_ * HW_);
    int v   = rem / HW_;
    int p   = rem - v * HW_;
    featT[((size_t)b * HW_ + p) * V_ + v] = (_Float16)feat[idx];
}

// ------------------- per (b,n): S[v]=sum gt*feat, gtsum, gtf2=sum gt*|feat|^2
__global__ __launch_bounds__(256) void stats_kernel(const float* __restrict__ feat,
                                                    const float* __restrict__ gt,
                                                    float* __restrict__ S,
                                                    float* __restrict__ gtsumw,
                                                    float* __restrict__ gtf2w) {
    int bn = blockIdx.x;
    int b  = bn >> 4;
    const float* gtp = gt + (size_t)bn * HW_;
    const float* fb  = feat + (size_t)b * V_ * HW_;

    float loc[V_];
#pragma unroll
    for (int v = 0; v < V_; v++) loc[v] = 0.f;
    float lg = 0.f, lf2 = 0.f;

    for (int p = threadIdx.x; p < HW_; p += 256) {
        float g = gtp[p];
        if (g != 0.f) {
            lg += g;
            float f2 = 0.f;
#pragma unroll
            for (int v = 0; v < V_; v++) {
                float fv = fb[(size_t)v * HW_ + p];
                loc[v] += g * fv;
                f2 += fv * fv;
            }
            lf2 += g * f2;
        }
    }

    __shared__ float part[V_ + 2][8];
    int lane = threadIdx.x & 31, wv = threadIdx.x >> 5;
#pragma unroll
    for (int q = 0; q < V_; q++) {
        float vq = loc[q];
        for (int m = 16; m >= 1; m >>= 1) vq += __shfl_xor(vq, m, 32);
        if (lane == 0) part[q][wv] = vq;
    }
    for (int m = 16; m >= 1; m >>= 1) lg += __shfl_xor(lg, m, 32);
    if (lane == 0) part[V_][wv] = lg;
    for (int m = 16; m >= 1; m >>= 1) lf2 += __shfl_xor(lf2, m, 32);
    if (lane == 0) part[V_ + 1][wv] = lf2;
    __syncthreads();

    if (threadIdx.x < V_ + 2) {
        float s = 0.f;
#pragma unroll
        for (int w = 0; w < 8; w++) s += part[threadIdx.x][w];
        if (threadIdx.x < V_)        S[(size_t)bn * V_ + threadIdx.x] = s;
        else if (threadIdx.x == V_)  gtsumw[bn] = s;
        else                         gtf2w[bn] = s;
    }
}

// ---------------- top-7 of pred*gt per (b,n) via 7 masked argmax passes
__global__ __launch_bounds__(256) void topk_kernel(const float* __restrict__ pred,
                                                   const float* __restrict__ gt,
                                                   int* __restrict__ idxw) {
    int bn = blockIdx.x;
    int b  = bn >> 4;
    const float* pm  = pred + (size_t)b * HW_;
    const float* gtp = gt + (size_t)bn * HW_;

    __shared__ float bv[256];
    __shared__ int   bi[256];
    __shared__ int   chosen[TOPK];

    for (int it = 0; it < TOPK; ++it) {
        float best = -3.4e38f;
        int   bidx = 0x7fffffff;
        for (int p = threadIdx.x; p < HW_; p += 256) {
            bool skip = false;
            for (int j = 0; j < it; j++)
                if (chosen[j] == p) skip = true;
            if (skip) continue;
            float w = pm[p] * gtp[p];
            if (w > best || (w == best && p < bidx)) { best = w; bidx = p; }
        }
        bv[threadIdx.x] = best;
        bi[threadIdx.x] = bidx;
        __syncthreads();
        for (int s = 128; s > 0; s >>= 1) {
            if (threadIdx.x < s) {
                float ov = bv[threadIdx.x + s];
                int   oi = bi[threadIdx.x + s];
                if (ov > bv[threadIdx.x] ||
                    (ov == bv[threadIdx.x] && oi < bi[threadIdx.x])) {
                    bv[threadIdx.x] = ov;
                    bi[threadIdx.x] = oi;
                }
            }
            __syncthreads();
        }
        if (threadIdx.x == 0) chosen[it] = bi[0];
        __syncthreads();
    }
    if (threadIdx.x < TOPK) idxw[bn * TOPK + threadIdx.x] = chosen[threadIdx.x];
}

// --- build 128 f16 rows per batch (16 mean rows + 112 gathered) + closed-form L2
__global__ __launch_bounds__(128) void rows_kernel(const float* __restrict__ feat,
                                                   const int* __restrict__ gmask,
                                                   const float* __restrict__ S,
                                                   const float* __restrict__ gtsumw,
                                                   const float* __restrict__ gtf2w,
                                                   const int* __restrict__ idxw,
                                                   _Float16* __restrict__ rows,
                                                   float* __restrict__ out) {
    int b = blockIdx.x, t = threadIdx.x;
    if (t < N_) {
        int bn  = b * N_ + t;
        float gs  = gtsumw[bn];
        float pix = fmaxf(gs, 1.0f);
        float m2  = 0.f;
#pragma unroll
        for (int v = 0; v < V_; v++) {
            float mv = S[(size_t)bn * V_ + v] / pix;
            m2 += mv * mv;
            rows[((size_t)b * 128 + t) * V_ + v] = (_Float16)mv;
        }
        // sum_p gt*(|f|^2 - 2 m.f + |m|^2) / pix  with  S = pix*m
        float l2 = (gtf2w[bn] - 2.f * pix * m2 + gs * m2) / pix;
        atomicAdd(out + 2, l2 * (float)gmask[bn]);
    } else {
        int k = t - N_;
        int n = k / TOPK;
        int j = k - n * TOPK;
        int p = idxw[(b * N_ + n) * TOPK + j];
        const float* fb = feat + (size_t)b * V_ * HW_;
#pragma unroll
        for (int v = 0; v < V_; v++)
            rows[((size_t)b * 128 + t) * V_ + v] = (_Float16)fb[(size_t)v * HW_ + p];
    }
}

// -------- WMMA sweep: logits = rows(128x32) x feat(32xHW), fused sigmoid+dice sums
__global__ __launch_bounds__(256) void dice_wmma_kernel(const _Float16* __restrict__ featT,
                                                        const _Float16* __restrict__ rows,
                                                        const float* __restrict__ gt,
                                                        float* __restrict__ sp2a,
                                                        float* __restrict__ tpa) {
    int b      = blockIdx.y;
    int chunk  = blockIdx.x;
    int lane   = threadIdx.x & 31;
    int rb     = threadIdx.x >> 5;      // row block 0..7  (rows rb*16 .. rb*16+15)
    int laneLo = lane & 15;
    int laneHi = lane >> 4;
    int cbase  = chunk * (TILES_PER_CHUNK * 16);   // first pixel of this chunk

    // A fragment: 16x32 f16.  Lane half selects K {0..15} / {16..31}.
    union { v16h v; v8h h[2]; } A;
    const _Float16* rp = rows + ((size_t)b * 128 + rb * 16 + laneLo) * V_ + laneHi * 16;
    A.v = *(const v16h*)(rp);

    // gt base pointer per accumulator register (chunk + pixel lane folded in)
    const float* gtp[8];
#pragma unroll
    for (int r = 0; r < 8; r++) {
        int row = rb * 16 + laneHi * 8 + r;
        int n   = (row < 16) ? row : (row - 16) / TOPK;
        gtp[r]  = gt + ((size_t)b * N_ + n) * HW_ + cbase + laneLo;
    }

    // B base pointer: 32B contiguous per lane from featT[b][p][v]
    const _Float16* bp = featT + ((size_t)b * HW_ + cbase + laneLo) * V_ + laneHi * 16;

    float sp2l[8], tpl[8];
#pragma unroll
    for (int r = 0; r < 8; r++) { sp2l[r] = 0.f; tpl[r] = 0.f; }

    // Fully unrolled: all tile addresses become immediate offsets
    // (B tile stride = 16*V_ halves = 1024 B, gt tile stride = 16 floats = 64 B),
    // and the scheduler can interleave next-tile loads with this tile's epilogue.
#pragma unroll
    for (int ti = 0; ti < TILES_PER_CHUNK; ++ti) {
        v16h Bv = *(const v16h*)(bp + ti * (16 * V_));

        v8f c = {};
        c = __builtin_amdgcn_wmma_f32_16x16x32_f16(
            /*neg_a=*/false, A.v, /*neg_b=*/false, Bv,
            /*c_mod=*/(short)0, c, /*reuse_a=*/false, /*reuse_b=*/false);

        // issue the 8 gt loads before touching c: they fill the WMMA hazard slots
        float tv[8];
#pragma unroll
        for (int r = 0; r < 8; r++) tv[r] = gtp[r][ti * 16];

#pragma unroll
        for (int r = 0; r < 8; r++) {
            // sigmoid via hardware v_exp + v_rcp (no IEEE div expansion)
            float pv = __builtin_amdgcn_rcpf(1.f + __expf(-c[r]));
            sp2l[r] += pv * pv;
            tpl[r]  += pv * tv[r];
        }
    }

    // reduce over the 16 pixel lanes within each half-wave, then accumulate
#pragma unroll
    for (int r = 0; r < 8; r++) {
        float s = sp2l[r], t = tpl[r];
        for (int m = 8; m >= 1; m >>= 1) {
            s += __shfl_xor(s, m, 32);
            t += __shfl_xor(t, m, 32);
        }
        if (laneLo == 0) {
            int row = rb * 16 + laneHi * 8 + r;
            atomicAdd(&sp2a[b * 128 + row], s);
            atomicAdd(&tpa[b * 128 + row], t);
        }
    }
}

// ---------------------------------------------------------------- finalize
__global__ void finalize_kernel(const float* __restrict__ sp2a,
                                const float* __restrict__ tpa,
                                const float* __restrict__ gtsumw,
                                const int* __restrict__ gmask,
                                float* __restrict__ out) {
    for (int i = threadIdx.x; i < B_ * 128; i += blockDim.x) {
        int b   = i >> 7;
        int row = i & 127;
        int n   = (row < 16) ? row : (row - 16) / TOPK;
        int bn  = b * N_ + n;
        float g     = gtsumw[bn];               // sum(gt^2) == sum(gt) for binary gt
        float denom = fmaxf(sp2a[i] + g, EPS_DICE);
        float dice  = 1.f - 2.f * tpa[i] / denom;
        float contrib = dice * (float)gmask[bn];
        atomicAdd(out + ((row < 16) ? 0 : 1), contrib);
    }
}

// ---------------------------------------------------------------- launch
extern "C" void kernel_launch(void* const* d_in, const int* in_sizes, int n_in,
                              void* d_out, int out_size, void* d_ws, size_t ws_size,
                              hipStream_t stream) {
    const float* pred  = (const float*)d_in[0];   // (4,1,160,160)
    const float* feat  = (const float*)d_in[1];   // (4,32,160,160)
    const float* gt    = (const float*)d_in[2];   // (4,16,160,160)
    const int*   gmask = (const int*)d_in[3];     // (4,16)
    float* out = (float*)d_out;                   // 3 floats

    float* ws_f = (float*)d_ws;
    float* S      = ws_f;                  // 2048
    float* gtsumw = ws_f + 2048;           // 64
    float* gtf2w  = ws_f + 2112;           // 64
    float* sp2a   = ws_f + 2176;           // 512
    float* tpa    = ws_f + 2688;           // 512  (contiguous with sp2a)
    int*   idxw   = (int*)(ws_f + 3200);   // 448
    _Float16* rowsw = (_Float16*)((char*)d_ws + 16384);   // 4*128*32 halves
    _Float16* featT = (_Float16*)((char*)d_ws + 65536);   // 4*25600*32 halves

    zero_kernel<<<4, 256, 0, stream>>>(out, sp2a, 1024);

    const int total = B_ * V_ * HW_;
    convert_kernel<<<(total + 255) / 256, 256, 0, stream>>>(feat, featT);

    stats_kernel<<<B_ * N_, 256, 0, stream>>>(feat, gt, S, gtsumw, gtf2w);
    topk_kernel<<<B_ * N_, 256, 0, stream>>>(pred, gt, idxw);
    rows_kernel<<<B_, 128, 0, stream>>>(feat, gmask, S, gtsumw, gtf2w, idxw, rowsw, out);

    dice_wmma_kernel<<<dim3(CHUNKS, B_), 256, 0, stream>>>(featT, rowsw, gt, sp2a, tpa);

    finalize_kernel<<<1, 256, 0, stream>>>(sp2a, tpa, gtsumw, gmask, out);
}
